// ChebConv_2834678415937
// MI455X (gfx1250) — compile-verified, matrix operand order checked by reading, not run on hardware
//
#include <hip/hip_runtime.h>
#include <hip/hip_bf16.h>
#include <math.h>

#define NN 50000

typedef __bf16 bf16_t;
typedef __attribute__((ext_vector_type(16))) __bf16 v16bf;
typedef __attribute__((ext_vector_type(8)))  __bf16 v8bf;
typedef __attribute__((ext_vector_type(8)))  float  v8f;

// ---------------- utility kernels ----------------

__global__ void zero_f32_kernel(float* p, int n) {
    int i = blockIdx.x * blockDim.x + threadIdx.x;
    if (i < n) p[i] = 0.0f;
}

__global__ void deg_kernel(const int* __restrict__ dst, float* __restrict__ deg, int E) {
    int e = blockIdx.x * blockDim.x + threadIdx.x;
    if (e < E) atomicAdd(&deg[dst[e]], 1.0f);
}

__global__ void dinv_kernel(const float* __restrict__ deg, float* __restrict__ dinv, int n) {
    int i = blockIdx.x * blockDim.x + threadIdx.x;
    if (i < n) {
        float d = deg[i];
        dinv[i] = (d > 0.0f) ? rsqrtf(fmaxf(d, 1.0f)) : 0.0f;
    }
}

// W: K x N (f32, row-major)  ->  Wt: N x K (bf16, row-major)
__global__ void cvt_wt_kernel(const float* __restrict__ W, bf16_t* __restrict__ Wt,
                              int K, int N) {
    int i = blockIdx.x * blockDim.x + threadIdx.x;
    if (i >= K * N) return;
    int k = i / N, n = i - k * N;
    Wt[(size_t)n * K + k] = (bf16_t)W[(size_t)k * N + n];
}

// x (f32, M x F) -> left half of Acat (bf16, row stride lda)
__global__ void cvt_x_kernel(const float* __restrict__ x, bf16_t* __restrict__ Acat,
                             int lda, int F, int total, int fshift) {
    int i = blockIdx.x * blockDim.x + threadIdx.x;
    if (i >= total) return;
    int node = i >> fshift, f = i & (F - 1);
    Acat[(size_t)node * lda + f] = (bf16_t)x[(size_t)node * F + f];
}

// agg[dst] += dinv[src] * H[src]   (H = bf16 left half of Acat)
__global__ void scatter_kernel(const int* __restrict__ src, const int* __restrict__ dst,
                               const bf16_t* __restrict__ H, int ldh, int F,
                               const float* __restrict__ dinv, float* __restrict__ agg,
                               int total, int gshift) {
    int i = blockIdx.x * blockDim.x + threadIdx.x;
    if (i >= total) return;
    int e = i >> gshift;                 // edge
    int g = i & ((1 << gshift) - 1);     // 4-feature group
    int s = src[e], d = dst[e];
    float w = dinv[s];
    const bf16_t* hp = H + (size_t)s * ldh + g * 4;
    float* ap = agg + (size_t)d * F + g * 4;
#pragma unroll
    for (int j = 0; j < 4; ++j) atomicAdd(ap + j, w * (float)hp[j]);
}

// Acat[:, F + f] = bf16( -agg * dinv )
__global__ void build_x1_kernel(const float* __restrict__ agg, const float* __restrict__ dinv,
                                bf16_t* __restrict__ Acat, int lda, int F,
                                int total, int fshift) {
    int i = blockIdx.x * blockDim.x + threadIdx.x;
    if (i >= total) return;
    int node = i >> fshift, f = i & (F - 1);
    Acat[(size_t)node * lda + F + f] = (bf16_t)(-agg[(size_t)node * F + f] * dinv[node]);
}

// ---------------- WMMA GEMM ----------------
// D = A(M x K, bf16, stride lda) @ Wt^T (Wt is N x K bf16) + bias, optional tanh.
// Each wave computes a 16x64 strip. 8 waves / block. N % 64 == 0, K % 32 == 0, M % 16 == 0.
__global__ void gemm_wmma_kernel(const bf16_t* __restrict__ A, int lda,
                                 const bf16_t* __restrict__ Wt,
                                 const float* __restrict__ bias,
                                 int M, int K, int N,
                                 bf16_t* __restrict__ outb, int ldo_b,
                                 float* __restrict__ outf, int ldo_f,
                                 int apply_tanh) {
    const int lane = threadIdx.x & 31;
    const int wave = threadIdx.x >> 5;
    const int m_tile = blockIdx.x * (blockDim.x >> 5) + wave;
    if (m_tile * 16 >= M) return;                 // wave-uniform: EXEC stays all-ones
    const int n_base = blockIdx.y * 64;

    const int half = lane >> 4;                   // 0: k 0-7 and 16-23, 1: k 8-15 and 24-31
    const int mr   = lane & 15;
    const int row  = m_tile * 16 + mr;
    const int kb   = half * 8;

    v8f acc[4];
#pragma unroll
    for (int t = 0; t < 4; ++t) acc[t] = (v8f){0.f,0.f,0.f,0.f,0.f,0.f,0.f,0.f};

    const bf16_t* arow = A + (size_t)row * lda;

    for (int k = 0; k < K; k += 32) {
        // A fragment: ISA 16-bit 16x32 layout
        v16bf afrag;
        {
            const v8bf lo = *(const v8bf*)(arow + k + kb);
            const v8bf hi = *(const v8bf*)(arow + k + 16 + kb);
#pragma unroll
            for (int j = 0; j < 8; ++j) { afrag[j] = lo[j]; afrag[8 + j] = hi[j]; }
        }
#pragma unroll
        for (int t = 0; t < 4; ++t) {
            const bf16_t* wrow = Wt + (size_t)(n_base + t * 16 + mr) * K;
            const v8bf lo = *(const v8bf*)(wrow + k + kb);
            const v8bf hi = *(const v8bf*)(wrow + k + 16 + kb);
            v16bf bfrag;
#pragma unroll
            for (int j = 0; j < 8; ++j) { bfrag[j] = lo[j]; bfrag[8 + j] = hi[j]; }
            acc[t] = __builtin_amdgcn_wmma_f32_16x16x32_bf16(
                false, afrag, false, bfrag, (short)0, acc[t], false, false);
        }
    }

    // D layout: element r -> row = half*8 + r, col = lane&15
#pragma unroll
    for (int t = 0; t < 4; ++t) {
        const int col = n_base + t * 16 + mr;
        const float bv = bias[col];
#pragma unroll
        for (int r = 0; r < 8; ++r) {
            const int m = m_tile * 16 + half * 8 + r;
            float v = acc[t][r] + bv;
            if (apply_tanh) v = tanhf(v);
            if (outb) outb[(size_t)m * ldo_b + col] = (bf16_t)v;
            else      outf[(size_t)m * ldo_f + col] = v;
        }
    }
}

// ---------------- host side ----------------

static inline int cdiv(int a, int b) { return (a + b - 1) / b; }

extern "C" void kernel_launch(void* const* d_in, const int* in_sizes, int n_in,
                              void* d_out, int out_size, void* d_ws, size_t ws_size,
                              hipStream_t stream) {
    const float* x   = (const float*)d_in[0];
    const int*   src = (const int*)  d_in[1];
    const int*   dst = (const int*)  d_in[2];
    const float* W1  = (const float*)d_in[3];
    const float* b1  = (const float*)d_in[4];
    const float* W2  = (const float*)d_in[5];
    const float* b2  = (const float*)d_in[6];
    const float* W3  = (const float*)d_in[7];
    const float* b3  = (const float*)d_in[8];
    const int E = in_sizes[1];

    // workspace carve-out (256B aligned slices)
    char* ws = (char*)d_ws;
    size_t off = 0;
    auto carve = [&](size_t bytes) -> void* {
        void* p = ws + off;
        off = (off + bytes + 255) & ~(size_t)255;
        return p;
    };
    bf16_t* AcatA = (bf16_t*)carve((size_t)NN * 512 * sizeof(bf16_t));
    bf16_t* AcatB = (bf16_t*)carve((size_t)NN * 512 * sizeof(bf16_t));
    float*  agg   = (float*) carve((size_t)NN * 256 * sizeof(float));
    float*  deg   = (float*) carve((size_t)NN * sizeof(float));
    float*  dinv  = (float*) carve((size_t)NN * sizeof(float));
    bf16_t* Wt1   = (bf16_t*)carve((size_t)256 * 256 * sizeof(bf16_t));
    bf16_t* Wt2   = (bf16_t*)carve((size_t)256 * 512 * sizeof(bf16_t));
    bf16_t* Wt3   = (bf16_t*)carve((size_t)128 * 512 * sizeof(bf16_t));

    const int B = 256;

    // degrees -> dinv
    zero_f32_kernel<<<cdiv(NN, B), B, 0, stream>>>(deg, NN);
    deg_kernel<<<cdiv(E, B), B, 0, stream>>>(dst, deg, E);
    dinv_kernel<<<cdiv(NN, B), B, 0, stream>>>(deg, dinv, NN);

    // weights -> bf16 transposed
    cvt_wt_kernel<<<cdiv(256 * 256, B), B, 0, stream>>>(W1, Wt1, 256, 256);
    cvt_wt_kernel<<<cdiv(512 * 256, B), B, 0, stream>>>(W2, Wt2, 512, 256);
    cvt_wt_kernel<<<cdiv(512 * 128, B), B, 0, stream>>>(W3, Wt3, 512, 128);

    // x -> left half of AcatA (stride 512)
    cvt_x_kernel<<<cdiv(NN * 128, B), B, 0, stream>>>(x, AcatA, 512, 128, NN * 128, 7);

    // one Cheb layer: aggregate from Acat(left F_in cols), build x1 into right half,
    // then GEMM concat @ Wt + bias.
    auto run_layer = [&](bf16_t* Acat, int F_in, int gshift, int fshift,
                         const bf16_t* Wt, const float* bias, int Nout,
                         bf16_t* outb, float* outf, int tanh_flag) {
        const int Kcat = 2 * F_in;
        zero_f32_kernel<<<cdiv(NN * F_in, B), B, 0, stream>>>(agg, NN * F_in);
        const int stotal = E * (F_in >> 2);
        scatter_kernel<<<cdiv(stotal, B), B, 0, stream>>>(src, dst, Acat, 512, F_in,
                                                          dinv, agg, stotal, gshift);
        build_x1_kernel<<<cdiv(NN * F_in, B), B, 0, stream>>>(agg, dinv, Acat, 512, F_in,
                                                              NN * F_in, fshift);
        dim3 grid(cdiv(NN / 16, 8), Nout / 64);
        gemm_wmma_kernel<<<grid, 256, 0, stream>>>(Acat, 512, Wt, bias,
                                                   NN, Kcat, Nout,
                                                   outb, 512, outf, Nout, tanh_flag);
    };

    // layer 1: F_in=128 (gshift=5: 32 groups of 4; fshift=7), out 256 -> AcatB, tanh
    run_layer(AcatA, 128, 5, 7, Wt1, b1, 256, AcatB, nullptr, 1);
    // hidden layers: F_in=256 (gshift=6, fshift=8), out 256, tanh, ping-pong
    run_layer(AcatB, 256, 6, 8, Wt2, b2, 256, AcatA, nullptr, 1);
    run_layer(AcatA, 256, 6, 8, Wt2, b2, 256, AcatB, nullptr, 1);
    run_layer(AcatB, 256, 6, 8, Wt2, b2, 256, AcatA, nullptr, 1);
    // output layer: F_in=256, out 128, no tanh, f32 -> d_out
    run_layer(AcatA, 256, 6, 8, Wt3, b3, 128, nullptr, (float*)d_out, 0);

    (void)n_in; (void)out_size; (void)ws_size;
}
